// Generator_20074677141536
// MI455X (gfx1250) — matrix operand order repaired and checked by reading, not compiled, not run on hardware
//
#include <hip/hip_runtime.h>

// Problem dims (fixed by the reference)
#define B_ 8
#define T_ 64
#define S_ 512
#define D_ 512
#define V_ 50000
#define M_ (B_ * T_)   // 512 rows of x

typedef __attribute__((ext_vector_type(16))) __bf16 v16bf;
typedef __attribute__((ext_vector_type(8)))  float  v8f;

// fp32 -> bf16 bits with round-to-nearest-even
__device__ __forceinline__ unsigned short f2bf(float f) {
  unsigned u = __float_as_uint(f);
  unsigned r = u + 0x7FFFu + ((u >> 16) & 1u);
  return (unsigned short)(r >> 16);
}

// Assemble a 16-element bf16 fragment from two 16B LDS chunks, `gap` ushorts apart
__device__ __forceinline__ v16bf ld_frag2(const unsigned short* p, int gap) {
  union { uint4 u[2]; v16bf v; } c;
  c.u[0] = *(const uint4*)(p);
  c.u[1] = *(const uint4*)(p + gap);
  return c.v;
}

// ---------------------------------------------------------------------------
// Kernel 1: alpha = sigmoid(x . W_pgen + b_pgen); attn_ = (1-alpha)*softmax(attn_dist)
// one block (256 thr) per (b,t) row
// ---------------------------------------------------------------------------
__global__ __launch_bounds__(256) void alpha_attn_kernel(
    const float* __restrict__ x, const float* __restrict__ attn_dist,
    const float* __restrict__ Wpg, const float* __restrict__ bpg,
    float* __restrict__ alpha, float* __restrict__ attnw)
{
  __shared__ float red[256];
  const int m = blockIdx.x;
  const int tid = threadIdx.x;

  // dot(x[m,:], W_pgen) over D_=512
  float p = x[(size_t)m * D_ + tid] * Wpg[tid]
          + x[(size_t)m * D_ + tid + 256] * Wpg[tid + 256];
  red[tid] = p; __syncthreads();
  for (int s = 128; s > 0; s >>= 1) { if (tid < s) red[tid] += red[tid + s]; __syncthreads(); }
  const float a = 1.f / (1.f + __expf(-(red[0] + bpg[0])));
  __syncthreads();
  if (tid == 0) alpha[m] = a;

  // softmax over S_=512
  const float v0 = attn_dist[(size_t)m * S_ + tid];
  const float v1 = attn_dist[(size_t)m * S_ + tid + 256];
  red[tid] = fmaxf(v0, v1); __syncthreads();
  for (int s = 128; s > 0; s >>= 1) { if (tid < s) red[tid] = fmaxf(red[tid], red[tid + s]); __syncthreads(); }
  const float mx = red[0]; __syncthreads();
  const float e0 = __expf(v0 - mx), e1 = __expf(v1 - mx);
  red[tid] = e0 + e1; __syncthreads();
  for (int s = 128; s > 0; s >>= 1) { if (tid < s) red[tid] += red[tid + s]; __syncthreads(); }
  const float scale = (1.f - a) / red[0];
  attnw[(size_t)m * S_ + tid]       = e0 * scale;
  attnw[(size_t)m * S_ + tid + 256] = e1 * scale;
}

// ---------------------------------------------------------------------------
// Kernel 2: logits = x @ W_proj + b_proj  via v_wmma_f32_16x16x32_bf16
// Block tile 64(M) x 64(N), K-step 32. 8 waves; wave w -> subtile (mt = w&3,
// nt = (w>>2)*2 .. +1) so the A fragment is reused across two WMMAs.
// Edge handling is branchless (V_ % 4 == 0 -> a float4 chunk is fully in or
// fully out of bounds); staging pointers are hoisted out of the K loop.
// ---------------------------------------------------------------------------
__global__ __launch_bounds__(256) void gemm_logits_kernel(
    const float* __restrict__ x, const float* __restrict__ Wp,
    const float* __restrict__ bias, float* __restrict__ logits)
{
  // rows padded to 40 ushorts (80B, multiple of 16B -> aligned uint4 loads)
  __shared__ alignas(16) unsigned short sA[64][40]; // [m][k] bf16
  __shared__ alignas(16) unsigned short sB[64][40]; // [n][k] bf16 (transposed tile)

  const int tid  = threadIdx.x;
  const int lane = tid & 31;
  const int wave = tid >> 5;
  const int m0 = blockIdx.y * 64;
  const int n0 = blockIdx.x * 64;

  const int mt = wave & 3;           // 0..3
  const int nt = (wave >> 2) << 1;   // 0 or 2 (covers nt, nt+1)

  // ---- loop-invariant staging coordinates -------------------------------
  // A tile: 64x32 floats/step; thread covers rows ar and ar+32, cols ac..ac+3
  const int ar = tid >> 3;
  const int ac = (tid & 7) << 2;
  const float* aptr0 = x + (size_t)(m0 + ar) * D_ + ac;
  const float* aptr1 = x + (size_t)(m0 + ar + 32) * D_ + ac;
  // B tile: 32(k)x64(n) floats/step; thread covers k rows kk, kk+16, cols gv..gv+3
  const int kk = tid >> 4;
  const int cn = (tid & 15) << 2;
  const int gv = n0 + cn;
  const bool bvalid = (gv < V_);               // whole float4 in or out
  const float* bptr0 = Wp + (size_t)kk * V_ + (bvalid ? gv : 0);
  const float* bptr1 = bptr0 + (size_t)16 * V_;

  // loop-invariant LDS store/load addresses
  unsigned short* sa0 = &sA[ar][ac];
  unsigned short* sa1 = &sA[ar + 32][ac];
  const unsigned short* pa = &sA[(mt << 4) + (lane & 15)][(lane >> 4) << 3];
  const unsigned short* pb = &sB[(nt << 4) + (lane & 15)][(lane >> 4) << 3];

  v8f acc0 = {};
  v8f acc1 = {};

  for (int k0 = 0; k0 < D_; k0 += 32) {
    // ---- stage A tile (fp32 -> bf16)
    const float4 fa0 = *(const float4*)aptr0;  aptr0 += 32;
    const float4 fa1 = *(const float4*)aptr1;  aptr1 += 32;
    sa0[0] = f2bf(fa0.x); sa0[1] = f2bf(fa0.y); sa0[2] = f2bf(fa0.z); sa0[3] = f2bf(fa0.w);
    sa1[0] = f2bf(fa1.x); sa1[1] = f2bf(fa1.y); sa1[2] = f2bf(fa1.z); sa1[3] = f2bf(fa1.w);

    // ---- stage B tile (fp32 -> bf16, transposed to [n][k]), branchless edge
    float4 fb0 = *(const float4*)bptr0;  bptr0 += (size_t)32 * V_;
    float4 fb1 = *(const float4*)bptr1;  bptr1 += (size_t)32 * V_;
    fb0.x = bvalid ? fb0.x : 0.f;  fb0.y = bvalid ? fb0.y : 0.f;
    fb0.z = bvalid ? fb0.z : 0.f;  fb0.w = bvalid ? fb0.w : 0.f;
    fb1.x = bvalid ? fb1.x : 0.f;  fb1.y = bvalid ? fb1.y : 0.f;
    fb1.z = bvalid ? fb1.z : 0.f;  fb1.w = bvalid ? fb1.w : 0.f;
    sB[cn + 0][kk]      = f2bf(fb0.x); sB[cn + 1][kk]      = f2bf(fb0.y);
    sB[cn + 2][kk]      = f2bf(fb0.z); sB[cn + 3][kk]      = f2bf(fb0.w);
    sB[cn + 0][kk + 16] = f2bf(fb1.x); sB[cn + 1][kk + 16] = f2bf(fb1.y);
    sB[cn + 2][kk + 16] = f2bf(fb1.z); sB[cn + 3][kk + 16] = f2bf(fb1.w);

    if (k0 + 32 < D_) { // hint next K-tile into cache (global_prefetch_b8)
      __builtin_prefetch(aptr0, 0, 0);
      __builtin_prefetch(bptr0, 0, 0);
    }
    __syncthreads();

    // ---- fragments per the CDNA5 16-bit WMMA VGPR layouts
    // A 16x32: lane m=l%16; elems 0..7 -> K=koff+0..7, 8..15 -> K=16+koff+0..7, koff=(l>=16)*8
    // B 32x16: lane n=l%16; elems 0..7 -> K=koff+0..7, 8..15 -> K=16+koff+0..7, koff=(l>=16)*8
    const v16bf a  = ld_frag2(pa, 16);
    const v16bf b0 = ld_frag2(pb, 16);
    const v16bf b1 = ld_frag2(pb + 16 * 40, 16); // subtile nt+1 (16 rows further)

    acc0 = __builtin_amdgcn_wmma_f32_16x16x32_bf16(false, a, false, b0, (short)0, acc0, false, false);
    acc1 = __builtin_amdgcn_wmma_f32_16x16x32_bf16(false, a, false, b1, (short)0, acc1, false, false);
    __syncthreads();
  }

  // ---- epilogue: C/D layout lane n=l%16, VGPR r -> M = r + 8*(l>=16); add bias
  // nc guards are uniform per 16-wide subtile (V_ % 16 == 0) -> scalar branch
  const int nc0 = n0 + (nt << 4) + (lane & 15);
  const int nc1 = nc0 + 16;
  const int mr  = m0 + (mt << 4) + ((lane >> 4) << 3);
  const float bia0 = (nc0 < V_) ? bias[nc0] : 0.f;
  const float bia1 = (nc1 < V_) ? bias[nc1] : 0.f;
#pragma unroll
  for (int r = 0; r < 8; ++r) {
    if (nc0 < V_) logits[(size_t)(mr + r) * V_ + nc0] = acc0[r] + bia0;
    if (nc1 < V_) logits[(size_t)(mr + r) * V_ + nc1] = acc1[r] + bia1;
  }
}

// ---------------------------------------------------------------------------
// Kernel 3: per-row max and sum(exp) over V (logits are L2-resident)
// ---------------------------------------------------------------------------
__global__ __launch_bounds__(256) void rowstats_kernel(
    const float* __restrict__ logits, float* __restrict__ rowmax, float* __restrict__ rowsum)
{
  __shared__ float red[256];
  const int m = blockIdx.x;
  const int tid = threadIdx.x;
  const float* row = logits + (size_t)m * V_;

  float mx = -3.4e38f;
  for (int v = tid; v < V_; v += 256) mx = fmaxf(mx, row[v]);
  red[tid] = mx; __syncthreads();
  for (int s = 128; s > 0; s >>= 1) { if (tid < s) red[tid] = fmaxf(red[tid], red[tid + s]); __syncthreads(); }
  mx = red[0]; __syncthreads();

  float sum = 0.f;
  for (int v = tid; v < V_; v += 256) sum += __expf(row[v] - mx);
  red[tid] = sum; __syncthreads();
  for (int s = 128; s > 0; s >>= 1) { if (tid < s) red[tid] += red[tid + s]; __syncthreads(); }
  if (tid == 0) { rowmax[m] = mx; rowsum[m] = red[0]; }
}

// ---------------------------------------------------------------------------
// Kernel 4: out = alpha[m] * softmax(logits) (in place)
// ---------------------------------------------------------------------------
__global__ __launch_bounds__(256) void normalize_kernel(
    float* __restrict__ out, const float* __restrict__ alpha,
    const float* __restrict__ rowmax, const float* __restrict__ rowsum)
{
  const int m = blockIdx.y;
  const int v = blockIdx.x * 256 + threadIdx.x;
  if (v < V_) {
    const size_t i = (size_t)m * V_ + v;
    out[i] = alpha[m] * __expf(out[i] - rowmax[m]) / rowsum[m];
  }
}

// ---------------------------------------------------------------------------
// Kernel 5: scatter-add attn mass into vocab dist (duplicates via atomics)
// ---------------------------------------------------------------------------
__global__ __launch_bounds__(256) void scatter_kernel(
    float* __restrict__ out, const long long* __restrict__ evo,
    const float* __restrict__ attnw)
{
  const int g  = blockIdx.x * 256 + threadIdx.x; // exactly B*T*S threads
  const int s  = g & (S_ - 1);
  const int bt = g >> 9;
  const int b  = bt / T_;
  const int v  = (int)evo[(size_t)b * S_ + s];
  atomicAdd(out + (size_t)bt * V_ + v, attnw[(size_t)bt * S_ + s]);
}

// ---------------------------------------------------------------------------
// Kernel 6: natural log in place
// ---------------------------------------------------------------------------
__global__ __launch_bounds__(256) void log_kernel(float* __restrict__ out)
{
  const int m = blockIdx.y;
  const int v = blockIdx.x * 256 + threadIdx.x;
  if (v < V_) {
    const size_t i = (size_t)m * V_ + v;
    out[i] = __logf(out[i]);
  }
}

extern "C" void kernel_launch(void* const* d_in, const int* in_sizes, int n_in,
                              void* d_out, int out_size, void* d_ws, size_t ws_size,
                              hipStream_t stream)
{
  (void)in_sizes; (void)n_in; (void)out_size; (void)ws_size;

  const float*     x    = (const float*)d_in[0];
  const float*     attn = (const float*)d_in[1];
  const long long* evo  = (const long long*)d_in[2]; // int64 indices
  const float*     Wp   = (const float*)d_in[3];
  const float*     bp   = (const float*)d_in[4];
  const float*     Wpg  = (const float*)d_in[5];
  const float*     bpg  = (const float*)d_in[6];
  float* out = (float*)d_out;

  // workspace layout (~1.03 MB): alpha | attn_ | rowmax | rowsum
  float* alpha  = (float*)d_ws;
  float* attnw  = alpha + M_;
  float* rowmax = attnw + (size_t)M_ * S_;
  float* rowsum = rowmax + M_;

  alpha_attn_kernel<<<M_, 256, 0, stream>>>(x, attn, Wpg, bpg, alpha, attnw);
  gemm_logits_kernel<<<dim3((V_ + 63) / 64, M_ / 64), 256, 0, stream>>>(x, Wp, bp, out);
  rowstats_kernel<<<M_, 256, 0, stream>>>(out, rowmax, rowsum);
  normalize_kernel<<<dim3((V_ + 255) / 256, M_), 256, 0, stream>>>(out, alpha, rowmax, rowsum);
  scatter_kernel<<<(B_ * T_ * S_) / 256, 256, 0, stream>>>(out, evo, attnw);
  log_kernel<<<dim3((V_ + 255) / 256, M_), 256, 0, stream>>>(out);
}
